// ABSA_Lstm_17437567221954
// MI455X (gfx1250) — matrix operand order repaired and checked by reading, not compiled, not run on hardware
//
#include <hip/hip_runtime.h>

// ---------------------------------------------------------------------------
// Masked LSTM (B=1024, T=80, D=H=300, C=3) for MI455X (gfx1250, wave32).
//
//  1) zero h/c state in workspace
//  2) pre-swizzle [Wx;Wh] (f32) into bf16 in a contiguous-K (k=16h+e) v_wmma B-operand lane
//     layout, padded: K 600->608 (19 K-tiles of 32), per-gate cols 300->304
//     (19 N-tiles of 16). Gate-major layout so one wave holds the i/f/g/o
//     accumulators for the same h-column tile and the gate loads are fixed
//     immediate offsets (19456 B apart) off one running 32-bit offset.
//  3) 80 sequential step kernels: gather x_t from emb (60 MB -> resident in
//     192 MB L2), build bf16 A=[x_t|h] in LDS, GEMM with
//     v_wmma_f32_16x16x32_bf16 (fp32 accum), fused fast-math
//     bias+sigmoid/tanh cell update with length masking.
//  4) final tiny projection h @ Wout + bout.
// ---------------------------------------------------------------------------

#define B_    1024
#define T_    80
#define D_    300
#define H_    300
#define KPAD  608      // (D_+H_) padded to 19*32
#define KT    19       // K tiles of 32
#define HT    19       // per-gate column tiles of 16 (304 cols, 300 real)
#define NT    (4*HT)   // 76 column tiles total (4 gates)

#define GATE_STRIDE_H  (HT * 512)   // halves between gates at same (kt,ht): 9728
#define KT_STRIDE_H    (NT * 512)   // halves between kt slabs: 38912

typedef __attribute__((ext_vector_type(8)))  __bf16 bf16x8;
typedef __attribute__((ext_vector_type(16))) __bf16 v16bf;
typedef __attribute__((ext_vector_type(8)))  float  v8f;

__device__ __forceinline__ unsigned short f2bf(float f) {
  unsigned int u = __float_as_uint(f);
  u += 0x7FFFu + ((u >> 16) & 1u);     // round-to-nearest-even
  return (unsigned short)(u >> 16);
}

__device__ __forceinline__ v16bf cat16(bf16x8 a, bf16x8 b) {
  return __builtin_shufflevector(a, b, 0,1,2,3,4,5,6,7,8,9,10,11,12,13,14,15);
}

// fast sigmoid/tanh: single v_exp_f32 + fast divide; tanh form is safe at +-inf
__device__ __forceinline__ float fsig(float x) {
  return __fdividef(1.0f, 1.0f + __expf(-x));
}
__device__ __forceinline__ float ftanh(float x) {
  return 1.0f - __fdividef(2.0f, 1.0f + __expf(2.0f * x));
}

// --- kernel 1: zero h and c state ------------------------------------------
__global__ void zero_state_k(float* __restrict__ Hst, float* __restrict__ Cst) {
  int i = blockIdx.x * 256 + threadIdx.x;           // 1200 blocks -> 307200
  if (i < B_ * H_) { Hst[i] = 0.0f; Cst[i] = 0.0f; }
}

// --- kernel 2: pre-swizzle weights into WMMA B-operand layout ---------------
// Flat element index = ((kt*NT + nt)*32 + lane)*16 + e
//   B layout (16-bit, 32x16 K-major tile): lane n=lane&15 is column,
//   K = kt*32 + (lane<16 ? 0 : 16) + e.
__global__ void build_wsw_k(const float* __restrict__ Wx,
                            const float* __restrict__ Wh,
                            unsigned short* __restrict__ Wsw) {
  int idx = blockIdx.x * 256 + threadIdx.x;         // 2888 blocks = 739328
  if (idx >= KT * NT * 512) return;
  int e    = idx & 15;
  int lane = (idx >> 4) & 31;
  int rest = idx >> 9;
  int nt   = rest % NT;
  int kt   = rest / NT;
  int K    = kt * 32 + ((lane < 16) ? 0 : 16) + e;
  int g    = nt / HT;                               // gate 0..3 = i,f,g,o
  int hc   = (nt % HT) * 16 + (lane & 15);          // padded per-gate column
  float w  = 0.0f;
  if (hc < H_) {
    int col = g * H_ + hc;                          // column in original 4H
    if (K < D_)            w = Wx[K * (4 * H_) + col];
    else if (K < D_ + H_)  w = Wh[(K - D_) * (4 * H_) + col];
    // K >= 600 -> zero pad
  }
  Wsw[idx] = f2bf(w);
}

// --- kernel 3: one fused LSTM time step ------------------------------------
// grid = 64 workgroups (16 batch rows each), block = 256 threads (8 waves).
__global__ void __launch_bounds__(256)
lstm_step_k(const int* __restrict__ sent, const int* __restrict__ lens,
            const float* __restrict__ emb, const float* __restrict__ bias,
            const unsigned short* __restrict__ Wsw,
            float* __restrict__ Hst, float* __restrict__ Cst, int t) {
  __shared__ __align__(16) unsigned short Ash[16 * KPAD];  // 19456 B
  __shared__ int tokS[16];
  __shared__ int mskS[16];

  const int row0 = blockIdx.x * 16;
  const int tid  = threadIdx.x;

  // Stage 0: per-row token + mask into LDS (one global read each).
  if (tid < 16) {
    int row   = row0 + tid;
    tokS[tid] = sent[row * T_ + t];
    mskS[tid] = (t < lens[row]) ? 1 : 0;
  }
  __syncthreads();

  // Stage 1: gather A = [ emb[tok] | h_row | 0-pad ] as bf16, row-major LDS.
  for (int idx = tid; idx < 16 * KPAD; idx += 256) {
    int m = idx / KPAD;
    int k = idx - m * KPAD;
    float v = 0.0f;
    if (k < D_) {
      v = emb[(size_t)tokS[m] * D_ + k];
    } else if (k < D_ + H_) {
      v = Hst[(row0 + m) * H_ + (k - D_)];
    }
    Ash[idx] = f2bf(v);
  }
  __syncthreads();

  const int lane   = tid & 31;
  const int wave   = tid >> 5;
  const int lr     = lane & 15;
  const int hiHalf = (lane >= 16);
  const int kbaseA = hiHalf ? 16 : 0;  // A layout: lanes 16..31 hold K+16 block
  const int madd   = hiHalf ? 8 : 0;   // D layout: lanes 16..31 hold M+8 rows
  const int m      = lr;               // A row for this lane

  for (int ht = wave; ht < HT; ht += 8) {
    v8f accI = {}, accF = {}, accG = {}, accO = {};

    // One running 32-bit offset; gates/halves are immediate-offset loads.
    unsigned off = (unsigned)ht * 512u + (unsigned)lane * 16u;  // halves
    const unsigned short* ash = &Ash[m * KPAD + kbaseA];

    for (int kt = 0; kt < KT; ++kt) {
      // A: two 16B chunks per lane (K..K+7, K+8..K+15) at kt*64B stride.
      const unsigned short* ap = ash + kt * 32;
      v16bf av = cat16(*(const bf16x8*)(ap), *(const bf16x8*)(ap + 8));

      const unsigned short* p = Wsw + off;
      v16bf bvI = cat16(*(const bf16x8*)(p),
                        *(const bf16x8*)(p + 8));
      v16bf bvF = cat16(*(const bf16x8*)(p + 1 * GATE_STRIDE_H),
                        *(const bf16x8*)(p + 1 * GATE_STRIDE_H + 8));
      v16bf bvG = cat16(*(const bf16x8*)(p + 2 * GATE_STRIDE_H),
                        *(const bf16x8*)(p + 2 * GATE_STRIDE_H + 8));
      v16bf bvO = cat16(*(const bf16x8*)(p + 3 * GATE_STRIDE_H),
                        *(const bf16x8*)(p + 3 * GATE_STRIDE_H + 8));

      accI = __builtin_amdgcn_wmma_f32_16x16x32_bf16(false, av, false, bvI, (short)0, accI, false, false);
      accF = __builtin_amdgcn_wmma_f32_16x16x32_bf16(false, av, false, bvF, (short)0, accF, false, false);
      accG = __builtin_amdgcn_wmma_f32_16x16x32_bf16(false, av, false, bvG, (short)0, accG, false, false);
      accO = __builtin_amdgcn_wmma_f32_16x16x32_bf16(false, av, false, bvO, (short)0, accO, false, false);

      off += KT_STRIDE_H;
    }

    // Fused cell update directly from accumulators.
    int hc = ht * 16 + lr;
    if (hc < H_) {
      float bI = bias[hc];
      float bF = bias[H_ + hc];
      float bG = bias[2 * H_ + hc];
      float bO = bias[3 * H_ + hc];
#pragma unroll
      for (int r = 0; r < 8; ++r) {
        int mrow = madd + r;
        if (mskS[mrow]) {                 // masked update: skip store past len
          int row  = row0 + mrow;
          float co = Cst[row * H_ + hc];
          float iv = fsig(accI[r] + bI);
          float fv = fsig(accF[r] + bF);
          float gv = ftanh(accG[r] + bG);
          float ov = fsig(accO[r] + bO);
          float cn = fv * co + iv * gv;
          float hn = ov * ftanh(cn);
          Cst[row * H_ + hc] = cn;
          Hst[row * H_ + hc] = hn;
        }
      }
    }
  }
}

// --- kernel 4: output projection h @ Wout + bout (1024x3) -------------------
__global__ void out_proj_k(const float* __restrict__ Hst,
                           const float* __restrict__ Wout,
                           const float* __restrict__ bout,
                           float* __restrict__ out) {
  int gid = blockIdx.x * 256 + threadIdx.x;
  if (gid >= B_ * 3) return;
  int row = gid / 3;
  int c   = gid - row * 3;
  float s = bout[c];
  for (int k = 0; k < H_; ++k) s += Hst[row * H_ + k] * Wout[k * 3 + c];
  out[gid] = s;
}

// ---------------------------------------------------------------------------
extern "C" void kernel_launch(void* const* d_in, const int* in_sizes, int n_in,
                              void* d_out, int out_size, void* d_ws, size_t ws_size,
                              hipStream_t stream) {
  const int*   sent = (const int*)d_in[0];
  const int*   lens = (const int*)d_in[1];
  const float* emb  = (const float*)d_in[2];
  const float* Wx   = (const float*)d_in[3];
  const float* Wh   = (const float*)d_in[4];
  const float* bias = (const float*)d_in[5];
  const float* Wout = (const float*)d_in[6];
  const float* bout = (const float*)d_in[7];
  float* out = (float*)d_out;

  // Workspace layout (256B aligned): h state | c state | swizzled bf16 weights
  char* ws = (char*)d_ws;
  float*          Hst = (float*)(ws);                       // 1,228,800 B
  float*          Cst = (float*)(ws + 1228800);             // 1,228,800 B
  unsigned short* Wsw = (unsigned short*)(ws + 2457600);    // 1,478,656 B

  zero_state_k<<<dim3(1200), dim3(256), 0, stream>>>(Hst, Cst);
  build_wsw_k<<<dim3(2888), dim3(256), 0, stream>>>(Wx, Wh, Wsw);

  for (int t = 0; t < T_; ++t) {
    lstm_step_k<<<dim3(B_ / 16), dim3(256), 0, stream>>>(
        sent, lens, emb, bias, Wsw, Hst, Cst, t);
  }

  out_proj_k<<<dim3(12), dim3(256), 0, stream>>>(Hst, Wout, bout, out);
}
